// FaceCategoryOutput_41016937677211
// MI455X (gfx1250) — matrix operand order repaired and checked by reading, not compiled, not run on hardware
//
#include <hip/hip_runtime.h>
#include <hip/hip_fp16.h>
#include <stdint.h>

typedef __attribute__((ext_vector_type(16))) _Float16 v16h;
typedef __attribute__((ext_vector_type(8)))  _Float16 v8h;
typedef __attribute__((ext_vector_type(8)))  float    v8f;

#define NCLS   100000
#define DIM    512
#define BATCH  512
#define NTILE  80
#define NBLKS  1250         // NCLS / NTILE
#define PSTRIDE 1280        // padded partial stride
#define SCALE_S 64.0f

// ---------------------------------------------------------------------------
// K1: embh[row][k] = f16( S * inputs[row][k] / ||inputs[row]|| )
// ---------------------------------------------------------------------------
__global__ void k_embh(const float* __restrict__ x, _Float16* __restrict__ embh) {
    const int row = blockIdx.x;
    const float* xr = x + (size_t)row * DIM;
    const float v0 = xr[threadIdx.x];
    const float v1 = xr[threadIdx.x + 256];
    float s = v0 * v0 + v1 * v1;
    #pragma unroll
    for (int m = 16; m >= 1; m >>= 1) s += __shfl_xor(s, m, 32);
    __shared__ float red[8];
    if ((threadIdx.x & 31) == 0) red[threadIdx.x >> 5] = s;
    __syncthreads();
    float tot = 0.f;
    #pragma unroll
    for (int i = 0; i < 8; ++i) tot += red[i];
    const float sc = SCALE_S * rsqrtf(tot);
    embh[(size_t)row * DIM + threadIdx.x]       = (_Float16)(v0 * sc);
    embh[(size_t)row * DIM + threadIdx.x + 256] = (_Float16)(v1 * sc);
}

// ---------------------------------------------------------------------------
// K2: whT[col][k] = f16( w[k][col] / ||w[:,col]|| )   (transposed, col-major)
// ---------------------------------------------------------------------------
__global__ void k_wnorm(const float* __restrict__ w, _Float16* __restrict__ whT) {
    const int j = blockIdx.x * 256 + threadIdx.x;
    if (j >= NCLS) return;
    float s = 0.f;
    #pragma unroll 4
    for (int k = 0; k < DIM; ++k) {
        const float v = w[(size_t)k * NCLS + j];
        s += v * v;
    }
    const float rinv = rsqrtf(s);
    _Float16* dst = whT + (size_t)j * DIM;
    #pragma unroll 1
    for (int k0 = 0; k0 < DIM; k0 += 8) {
        v8h h;
        #pragma unroll
        for (int u = 0; u < 8; ++u)
            h[u] = (_Float16)(w[(size_t)(k0 + u) * NCLS + j] * rinv);
        *(v8h*)(dst + k0) = h;   // 16B store
    }
}

// ---------------------------------------------------------------------------
// K3: fused GEMM + exp(x - S), per-block row partial sums.
//     Block: 512 threads = 16 waves. Block tile: M=512 (full), N=80.
//     K streamed in 32-deep slabs, double-buffered LDS, async global->LDS.
//     A fragments ping-pong between two register sets (no copy, no WAR nops).
// ---------------------------------------------------------------------------
__launch_bounds__(512, 1)
__global__ void k_gemm(const _Float16* __restrict__ embh,
                       const _Float16* __restrict__ whT,
                       float*          __restrict__ out,
                       float*          __restrict__ partial) {
    __shared__ _Float16 Bs[2][NTILE * 32];       // [buf][col][k], 5 KB each

    const int tid     = threadIdx.x;
    const int lane    = tid & 31;
    const int wave    = tid >> 5;                // 0..15
    const int mbase   = wave * 32;               // each wave: 32 rows
    const int colbase = blockIdx.x * NTILE;
    const int l16     = lane & 15;
    const int h       = lane >> 4;               // half-select
    const bool copier = (wave < 10);             // waves issuing 1 async per slab

    v8f acc[2][5] = {};                          // 2 M-subtiles x 5 N-subtiles

    const uint32_t lds_base = (uint32_t)(uintptr_t)(&Bs[0][0]);

    // issue one K-slab's async copies into buffer `buf` (320 x b128 chunks)
    auto stage = [&](int kslab, int buf) {
        if (tid < 320) {
            const _Float16* gbase = whT + (size_t)colbase * DIM + kslab * 32;
            const int col = tid >> 2, part = tid & 3;        // chunk = tid
            const uint32_t voff = (uint32_t)(col * (DIM * 2) + part * 16);
            const uint32_t ldsa = lds_base +
                (uint32_t)(buf * (NTILE * 32 * 2) + col * 64 + part * 16);
            asm volatile("global_load_async_to_lds_b128 %0, %1, %2 offset:0"
                         :: "v"(ldsa), "v"(voff), "s"(gbase) : "memory");
        }
    };

    // load both A fragments (16-bit A operand layout) for K-step kk
    auto load_a = [&](v16h a[2], int kk) {
        #pragma unroll
        for (int mt = 0; mt < 2; ++mt) {
            const _Float16* ar =
                embh + (size_t)(mbase + mt * 16 + l16) * DIM + kk * 32;
            const v8h lo = *(const v8h*)(ar + 16 * h);      // K0+16h..+7   -> v0-3
            const v8h hi = *(const v8h*)(ar + 16 * h + 8);  // K0+16h+8..  -> v4-7
            v16h t;
            #pragma unroll
            for (int u = 0; u < 8; ++u) { t[u] = lo[u]; t[u + 8] = hi[u]; }
            a[mt] = t;
        }
    };

    // one K-step: consume slab kk (buffer kk&1) with aCur, prefetch into aNext
    auto step = [&](int kk, bool last, v16h aCur[2], v16h aNext[2]) {
        if (!last) {
            stage(kk + 1, (kk + 1) & 1);         // overlap next slab with compute
            if (copier) asm volatile("s_wait_asynccnt 1" ::: "memory");
        } else {
            if (copier) asm volatile("s_wait_asynccnt 0" ::: "memory");
        }
        __syncthreads();                         // slab kk resident for all waves

        const _Float16* bufp = &Bs[kk & 1][0];
        v16h bfrag[5];                           // one dscnt wait, then WMMA chain
        #pragma unroll
        for (int nt = 0; nt < 5; ++nt)
            bfrag[nt] = *(const v16h*)(bufp + (nt * 16 + l16) * 32 + 16 * h);

        if (!last) load_a(aNext, kk + 1);        // prefetch next A during WMMAs

        #pragma unroll
        for (int nt = 0; nt < 5; ++nt) {
            acc[0][nt] = __builtin_amdgcn_wmma_f32_16x16x32_f16(
                false, aCur[0], false, bfrag[nt], (short)0, acc[0][nt], false, false);
            acc[1][nt] = __builtin_amdgcn_wmma_f32_16x16x32_f16(
                false, aCur[1], false, bfrag[nt], (short)0, acc[1][nt], false, false);
        }
        __syncthreads();                         // reads done before slab reuse
    };

    v16h afragA[2], afragB[2];
    stage(0, 0);                                 // prologue: slab 0 in flight
    load_a(afragA, 0);

    #pragma unroll 1
    for (int kk2 = 0; kk2 < 8; ++kk2) {          // 2 K-steps per iteration
        step(2 * kk2,     false,      afragA, afragB);
        step(2 * kk2 + 1, kk2 == 7,   afragB, afragA);
    }

    // ---- epilogue: exp(x - S) (shift-invariant softmax), partial row sums ----
    #pragma unroll
    for (int mt = 0; mt < 2; ++mt) {
        #pragma unroll
        for (int r = 0; r < 8; ++r) {
            const int row = mbase + mt * 16 + 8 * h + r;     // C/D layout: M=r+8h
            float rowsum = 0.f;
            #pragma unroll
            for (int nt = 0; nt < 5; ++nt) {
                const int col = colbase + nt * 16 + l16;     // C/D layout: N=lane%16
                const float e = __expf(acc[mt][nt][r] - SCALE_S);
                out[(size_t)row * NCLS + col] = e;
                rowsum += e;
            }
            #pragma unroll
            for (int m = 8; m >= 1; m >>= 1) rowsum += __shfl_xor(rowsum, m, 16);
            if (l16 == 0) partial[(size_t)row * PSTRIDE + blockIdx.x] = rowsum;
        }
    }
}

// ---------------------------------------------------------------------------
// K4: ArcFace margin fixup at (row, label[row]); records row-sum delta.
//     Recovers 64*cos = log(e)+64; e in [e^-77, e^-51] stays normal f32.
// ---------------------------------------------------------------------------
__global__ void k_fixup(const int* __restrict__ label, float* __restrict__ out,
                        float* __restrict__ fixdel) {
    const int row = blockIdx.x * 256 + threadIdx.x;
    if (row >= BATCH) return;
    const int lab = label[row];
    float* p = out + (size_t)row * NCLS + lab;
    const float e_old = *p;
    const float v  = logf(fmaxf(e_old, 1e-38f)) + SCALE_S;
    const float ct = fminf(fmaxf(v * (1.0f / SCALE_S), -1.f), 1.f);
    const float vn = cosf(acosf(ct) + 0.5f) * SCALE_S;
    const float e_new = __expf(vn - SCALE_S);
    *p = e_new;
    fixdel[row] = e_new - e_old;
}

// ---------------------------------------------------------------------------
// K5: deterministic per-row reduction of partial sums (+margin delta) -> 1/sum
// ---------------------------------------------------------------------------
__global__ void k_rowsum(const float* __restrict__ partial,
                         const float* __restrict__ fixdel,
                         float* __restrict__ rinv) {
    const int row = blockIdx.x;
    float s = 0.f;
    for (int i = threadIdx.x; i < NBLKS; i += 128)
        s += partial[(size_t)row * PSTRIDE + i];
    #pragma unroll
    for (int m = 16; m >= 1; m >>= 1) s += __shfl_xor(s, m, 32);
    __shared__ float red[4];
    if ((threadIdx.x & 31) == 0) red[threadIdx.x >> 5] = s;
    __syncthreads();
    if (threadIdx.x == 0)
        rinv[row] = 1.0f / (red[0] + red[1] + red[2] + red[3] + fixdel[row]);
}

// ---------------------------------------------------------------------------
// K6: out *= rinv[row]  (float4 streaming pass)
// ---------------------------------------------------------------------------
__global__ void k_norm(float* __restrict__ out, const float* __restrict__ rinv) {
    const size_t idx = (size_t)blockIdx.x * 256 + threadIdx.x;   // float4 index
    const int row = (int)(idx / (NCLS / 4));
    float4 v = ((float4*)out)[idx];
    const float r = rinv[row];
    v.x *= r; v.y *= r; v.z *= r; v.w *= r;
    ((float4*)out)[idx] = v;
}

// ---------------------------------------------------------------------------
extern "C" void kernel_launch(void* const* d_in, const int* in_sizes, int n_in,
                              void* d_out, int out_size, void* d_ws, size_t ws_size,
                              hipStream_t stream) {
    const float* x     = (const float*)d_in[0];
    const int*   label = (const int*)d_in[1];
    const float* w     = (const float*)d_in[2];
    float*       out   = (float*)d_out;
    char*        ws    = (char*)d_ws;

    // workspace layout (all 256B aligned)
    _Float16* whT     = (_Float16*)(ws);                 // 100000*512*2 = 102,400,000
    _Float16* embh    = (_Float16*)(ws + 102400000);     // 512*512*2    =     524,288
    float*    partial = (float*)(ws + 102924288);        // 512*1280*4   =   2,621,440
    float*    rinv    = (float*)(ws + 105545728);        // 512*4
    float*    fixdel  = (float*)(ws + 105547776);        // 512*4

    k_embh  <<<BATCH, 256, 0, stream>>>(x, embh);
    k_wnorm <<<(NCLS + 255) / 256, 256, 0, stream>>>(w, whT);
    k_gemm  <<<NBLKS, 512, 0, stream>>>(embh, whT, out, partial);
    k_fixup <<<2, 256, 0, stream>>>(label, out, fixdel);
    k_rowsum<<<BATCH, 128, 0, stream>>>(partial, fixdel, rinv);
    k_norm  <<<(BATCH * (NCLS / 4)) / 256, 256, 0, stream>>>(out, rinv);
}
